// LongTermROUNDModel_70703751627283
// MI455X (gfx1250) — matrix operand order repaired and checked by reading, not compiled, not run on hardware
//
#include <hip/hip_runtime.h>

// ---------------------------------------------------------------------------
// Types for CDNA5 WMMA (wave32) and TDM descriptors
// ---------------------------------------------------------------------------
typedef __attribute__((ext_vector_type(16))) __bf16   bf16x16;
typedef __attribute__((ext_vector_type(8)))  __bf16   bf16x8;
typedef __attribute__((ext_vector_type(8)))  float    floatx8;
typedef __attribute__((ext_vector_type(4)))  unsigned u32x4;
typedef __attribute__((ext_vector_type(8)))  int      i32x8;
typedef __attribute__((ext_vector_type(4)))  int      i32x4;

#define B_DIM 16
#define S_DIM 1024
#define D_IN  256
#define H_DIM 256
#define D_OUT 2048
#define NFEAT 7
#define K_RD  (H_DIM * NFEAT)   // 1792
#define M_DIM (B_DIM * S_DIM)   // 16384
#define WOBBLE_INC 0.015625f
#define LDS_PITCH 40            // 32 elems + 8 pad (80B rows, via TDM pad)

__device__ __forceinline__ __bf16 f2bf(float f) {
    unsigned u = __builtin_bit_cast(unsigned, f);
    u += 0x7FFFu + ((u >> 16) & 1u);          // round-to-nearest-even
    unsigned short s = (unsigned short)(u >> 16);
    return __builtin_bit_cast(__bf16, s);
}

__device__ __forceinline__ float wrap_pi(float p) {
    // atan2(sin(p), cos(p)) == p - 2*pi*rint(p/(2*pi))
    return p - 6.28318530717958647692f * rintf(p * 0.15915494309189533577f);
}

__device__ __forceinline__ bf16x16 make_afrag(const __bf16* base, int off) {
    bf16x8 alo = *(const bf16x8*)(base + off);
    bf16x8 ahi = *(const bf16x8*)(base + off + 16);
    return __builtin_shufflevector(alo, ahi,
        0,1,2,3,4,5,6,7,8,9,10,11,12,13,14,15);
}

// ---------------------------------------------------------------------------
// TDM: load a 2D tile (32 x 128 rows of bf16, row stride 1792 elems) into LDS
// with 16B padding after each 64B row (-> 80B LDS pitch). ISA §8.3/§8.4.
// This toolchain exposes the 6-arg builtin form (extra zero-filled group).
// ---------------------------------------------------------------------------
__device__ __forceinline__ void tdm_load_tile(unsigned lds_off, const void* gp) {
    unsigned long long ga = (unsigned long long)gp;
    u32x4 g0;
    g0[0] = 1u;                                      // count=1, user mode
    g0[1] = lds_off;                                 // LDS byte address
    g0[2] = (unsigned)(ga & 0xffffffffu);            // global_addr[31:0]
    g0[3] = (unsigned)((ga >> 32) & 0x1ffffffu)      // global_addr[56:32]
          | (2u << 30);                              // type=2 ("image")
    i32x8 g1;
    g1[0] = (1 << 16)                                // data_size: 2 bytes
          | (1 << 20)                                // pad_enable
          | (3 << 22)                                // pad_interval: 16 DWORDs
          | (3 << 25);                               // pad_amount: 4 DWORDs
    g1[1] = (32 & 0xffff) << 16;                     // tensor_dim0[15:0]=32
    g1[2] = (128 & 0xffff) << 16;                    // dim0 hi=0 | tensor_dim1 lo=128
    g1[3] = (32 << 16);                              // dim1 hi=0 | tile_dim0=32
    g1[4] = 128;                                     // tile_dim1=128, tile_dim2=0
    g1[5] = K_RD;                                    // tensor_dim0_stride = 1792
    g1[6] = 0;                                       // stride hi | dim1_stride lo
    g1[7] = 0;
    i32x4 z4 = {0, 0, 0, 0};                         // groups 2/3 unused (2D)
    i32x8 z8 = {0, 0, 0, 0, 0, 0, 0, 0};
    __builtin_amdgcn_tensor_load_to_lds(g0, g1, z4, z4, z8, 0);
}

__device__ __forceinline__ unsigned lds_offset_of(const void* p) {
    // generic shared pointer: low 32 bits are the LDS byte offset
    return (unsigned)(unsigned long long)p;
}

// ---------------------------------------------------------------------------
// fp32 -> bf16 conversion
// ---------------------------------------------------------------------------
__global__ void cvt_bf16_kernel(const float* __restrict__ src,
                                __bf16* __restrict__ dst, int n) {
    int i = blockIdx.x * blockDim.x + threadIdx.x;
    if (i < n) dst[i] = f2bf(src[i]);
}

// ---------------------------------------------------------------------------
// is_repeat[t] = (t>0) && all(x[:,t,:] == x[:,t-1,:])
// ---------------------------------------------------------------------------
__global__ __launch_bounds__(256) void flags_kernel(const float* __restrict__ x,
                                                    int* __restrict__ flags) {
    int t = blockIdx.x;
    int d = threadIdx.x;
    int ok = (t != 0);
    if (t != 0) {
        for (int b = 0; b < B_DIM; ++b) {
            size_t i = ((size_t)b * S_DIM + t) * D_IN + d;
            size_t j = ((size_t)b * S_DIM + (t - 1)) * D_IN + d;
            if (x[i] != x[j]) { ok = 0; break; }
        }
    }
    ok = __syncthreads_and(ok);
    if (threadIdx.x == 0) flags[t] = ok;
}

// ---------------------------------------------------------------------------
// Embed GEMM + phase-delta epilogue:
//   pt = x @ We^T + be ; dphi = omega * 0.5 * wrap(pt)
// M=16384, N=H=256, K=256. One 16x16 tile per wave, bf16 WMMA.
// dphi is written into the ph_hist output region (scan overwrites in place).
// ---------------------------------------------------------------------------
__global__ __launch_bounds__(256) void embed_kernel(
    const __bf16* __restrict__ xb, const __bf16* __restrict__ Web,
    const float* __restrict__ be, const float* __restrict__ omega,
    float* __restrict__ dphi)
{
    const int lane = threadIdx.x & 31;
    const int wave = threadIdx.x >> 5;
    const int m0   = blockIdx.x * 16;
    const int n0   = blockIdx.y * 128 + wave * 16;

    const int r    = lane & 15;
    const int half = lane >> 4;
    const int oA   = half * 8;
    const int kB   = half * 16;
    const int nB   = n0 + r;

    floatx8 acc = {};
    for (int k0 = 0; k0 < D_IN; k0 += 32) {
        bf16x16 a = make_afrag(xb + (size_t)(m0 + r) * D_IN + k0, oA);
        bf16x16 b = make_afrag(Web + (size_t)nB * D_IN + k0, oA);
        acc = __builtin_amdgcn_wmma_f32_16x16x32_bf16(
            false, a, false, b, (short)0, acc, false, false);
    }
    const float ben = be[nB];
    const float omn = omega[nB];
    #pragma unroll
    for (int i = 0; i < 8; ++i) {
        int mi = i + half * 8;
        float pt = acc[i] + ben;
        dphi[(size_t)(m0 + mi) * H_DIM + nB] = omn * 0.5f * wrap_pi(pt);
    }
}

// ---------------------------------------------------------------------------
// Sequential recurrence: 4096 independent (b,h) chains over t=0..1023.
// ph_io holds dphi on entry; overwritten with ph (read-before-write per idx).
// ---------------------------------------------------------------------------
__global__ __launch_bounds__(256) void scan_kernel(
    float* ph_io, const int* __restrict__ flags, float* __restrict__ wb_out)
{
    int u = blockIdx.x * 256 + threadIdx.x;
    int b = u >> 8;
    int h = u & 255;
    float ph = 0.f, wb = 0.f;
    size_t base = (size_t)b * S_DIM * H_DIM + h;
    for (int t = 0; t < S_DIM; ++t) {
        size_t idx = base + (size_t)t * H_DIM;
        float d = ph_io[idx];
        float wbe = wb + WOBBLE_INC;
        ph = ph + d - __sinf(wbe);
        wb = flags[t] ? (wbe + d) : wbe;
        ph_io[idx] = ph;
        wb_out[idx] = wb;
    }
}

// ---------------------------------------------------------------------------
// Feature matrix (one-time, bf16): feats[m, f*H + h]
// ---------------------------------------------------------------------------
__global__ __launch_bounds__(256) void feats_kernel(
    const float* __restrict__ ph, const float* __restrict__ wb,
    __bf16* __restrict__ feats)
{
    int i = blockIdx.x * 256 + threadIdx.x;
    int m = i >> 8;
    int h = i & 255;
    float p = ph[i];
    float w = wb[i];
    __bf16* o = feats + (size_t)m * K_RD + h;
    o[0 * H_DIM] = f2bf(__cosf(p));
    o[1 * H_DIM] = f2bf(__sinf(p));
    o[2 * H_DIM] = f2bf(__cosf(0.5f * p));
    o[3 * H_DIM] = f2bf(__sinf(0.5f * p));
    o[4 * H_DIM] = f2bf(__cosf(w));
    o[5 * H_DIM] = f2bf(__sinf(w));
    o[6 * H_DIM] = f2bf(p);
}

// ---------------------------------------------------------------------------
// Readout GEMM: logits = feats @ Wr^T + br.
// M=16384, N=2048, K=1792. 128x128 block tile, 256 threads (8 waves).
// Wave w owns 4 M-subtiles x 2 N-subtiles (8 fp32 accumulators).
// A/B K-slabs staged by the Tensor Data Mover, double-buffered in LDS:
// the TDM transfer of step k+1 overlaps the 8 WMMAs of step k.
// ---------------------------------------------------------------------------
__global__ __launch_bounds__(256) void readout_kernel(
    const __bf16* __restrict__ feats, const __bf16* __restrict__ Wrb,
    const float* __restrict__ br, float* __restrict__ logits)
{
    __shared__ __bf16 Alds[2][128 * LDS_PITCH + 8];   // +16B trailing-pad slack
    __shared__ __bf16 Blds[2][128 * LDS_PITCH + 8];
    const int tid  = threadIdx.x;
    const int lane = tid & 31;
    const int wave = tid >> 5;
    const int m0   = blockIdx.x * 128;
    const int n0   = blockIdx.y * 128;
    const int r    = lane & 15;
    const int half = lane >> 4;
    const int wm   = (wave >> 2) * 4;      // first of 4 M-subtiles (0 or 4)
    const int wn   = (wave & 3) * 2;       // first of 2 N-subtiles (0,2,4,6)

    const __bf16* Abase = feats + (size_t)m0 * K_RD;
    const __bf16* Bbase = Wrb   + (size_t)n0 * K_RD;

    // prologue: TDM-stage slab 0 into buffer 0
    if (wave == 0) tdm_load_tile(lds_offset_of(&Alds[0][0]), Abase);
    if (wave == 1) tdm_load_tile(lds_offset_of(&Blds[0][0]), Bbase);

    floatx8 acc[4][2] = {};
    int cur = 0;
    for (int k0 = 0; k0 < K_RD; k0 += 32) {
        __builtin_amdgcn_s_wait_tensorcnt(0);   // cnt==0 for non-issuing waves
        __syncthreads();                        // slab[cur] ready; buf[cur^1] free

        int kn = k0 + 32;
        if (kn < K_RD) {                        // overlap next DMA with compute
            if (wave == 0) tdm_load_tile(lds_offset_of(&Alds[cur ^ 1][0]), Abase + kn);
            if (wave == 1) tdm_load_tile(lds_offset_of(&Blds[cur ^ 1][0]), Bbase + kn);
        }

        const __bf16* As = &Alds[cur][0];
        const __bf16* Bs = &Blds[cur][0];
        bf16x16 bfrag[2];
        #pragma unroll
        for (int t = 0; t < 2; ++t) {
            int nn = (wn + t) * 16 + r;
            bfrag[t] = make_afrag(&Bs[nn * LDS_PITCH], half * 8);
        }
        #pragma unroll
        for (int j = 0; j < 4; ++j) {
            int rr = (wm + j) * 16 + r;
            bf16x16 a = make_afrag(&As[rr * LDS_PITCH], half * 8);
            acc[j][0] = __builtin_amdgcn_wmma_f32_16x16x32_bf16(
                false, a, false, bfrag[0], (short)0, acc[j][0], false, false);
            acc[j][1] = __builtin_amdgcn_wmma_f32_16x16x32_bf16(
                false, a, false, bfrag[1], (short)0, acc[j][1], false, false);
        }
        cur ^= 1;
    }
    #pragma unroll
    for (int t = 0; t < 2; ++t) {
        int ng = n0 + (wn + t) * 16 + r;
        float brn = br[ng];
        #pragma unroll
        for (int j = 0; j < 4; ++j) {
            #pragma unroll
            for (int i = 0; i < 8; ++i) {
                int mg = m0 + (wm + j) * 16 + i + half * 8;
                logits[(size_t)mg * D_OUT + ng] = acc[j][t][i] + brn;
            }
        }
    }
}

// ---------------------------------------------------------------------------
// Launch
// ---------------------------------------------------------------------------
extern "C" void kernel_launch(void* const* d_in, const int* in_sizes, int n_in,
                              void* d_out, int out_size, void* d_ws, size_t ws_size,
                              hipStream_t stream) {
    const float* x     = (const float*)d_in[0];   // [B,S,D_IN]
    const float* We    = (const float*)d_in[1];   // [H,D_IN]
    const float* be    = (const float*)d_in[2];   // [H]
    const float* omega = (const float*)d_in[3];   // [H]
    const float* Wr    = (const float*)d_in[4];   // [D_OUT, 7H]
    const float* br    = (const float*)d_in[5];   // [D_OUT]

    // workspace layout (~71.1 MB total, 256B-aligned offsets)
    char* ws = (char*)d_ws;
    __bf16* xb    = (__bf16*)(ws);                      //  8,388,608 B
    __bf16* Web   = (__bf16*)(ws +  8388608);           //    131,072 B
    __bf16* Wrb   = (__bf16*)(ws +  8519680);           //  7,340,032 B
    int*    flags = (int*)  (ws + 15859712);            //      4,096 B
    __bf16* feats = (__bf16*)(ws + 15863808);           // 58,720,256 B

    const int NX  = M_DIM * D_IN;        // 4,194,304
    const int NWE = H_DIM * D_IN;        //    65,536
    const int NWR = D_OUT * K_RD;        // 3,670,016

    cvt_bf16_kernel<<<(NX  + 255) / 256, 256, 0, stream>>>(x,  xb,  NX);
    cvt_bf16_kernel<<<(NWE + 255) / 256, 256, 0, stream>>>(We, Web, NWE);
    cvt_bf16_kernel<<<(NWR + 255) / 256, 256, 0, stream>>>(Wr, Wrb, NWR);
    flags_kernel<<<S_DIM, 256, 0, stream>>>(x, flags);

    float* out    = (float*)d_out;
    float* ph_out = out + (size_t)M_DIM * D_OUT;        // logits first
    float* wb_out = ph_out + (size_t)M_DIM * H_DIM;

    embed_kernel<<<dim3(M_DIM / 16, H_DIM / 128), 256, 0, stream>>>(
        xb, Web, be, omega, ph_out);
    scan_kernel<<<(B_DIM * H_DIM) / 256, 256, 0, stream>>>(
        ph_out, flags, wb_out);
    feats_kernel<<<(M_DIM * H_DIM) / 256, 256, 0, stream>>>(
        ph_out, wb_out, feats);

    readout_kernel<<<dim3(M_DIM / 128, D_OUT / 128), 256, 0, stream>>>(
        feats, Wrb, br, out);
}
